// Decoder_66383014527727
// MI455X (gfx1250) — matrix operand order repaired and checked by reading, not compiled, not run on hardware
//
#include <hip/hip_runtime.h>
#include <hip/hip_bf16.h>
#include <stdint.h>

// ---------------------------------------------------------------------------
// MI455X (gfx1250) persistent attention-decoder.
//
// Roofline: 600 sequential steps x ~250 MFLOP — FLOP-trivial for WMMA, the
// working set (14 MB f16 weights + 39 MB memory/pitch tensors) is L2-resident
// (192 MB), so the kernel is L2-bandwidth + sequential-latency bound.
// Persistent grid: 16 WGs x 256 threads = 128 wave32, atomic grid barrier,
// every dense GEMM via v_wmma_f32_16x16x32_f16 (batch B=16 == WMMA M=16:
// one wave owns one 16x16 output tile). Dual WMMA accumulators break the
// D->C serial chain so the unrolled loop body co-executes instead of v_nop.
// ---------------------------------------------------------------------------

typedef __attribute__((ext_vector_type(16))) _Float16 v16h;
typedef __attribute__((ext_vector_type(8)))  _Float16 h8;
typedef __attribute__((ext_vector_type(4)))  _Float16 h4;
typedef __attribute__((ext_vector_type(8)))  float    v8f;

#define NWG   16
#define NTHR  256
#define NTOT  (NWG * NTHR)
#define TDEC  600
#define TENC  600
#define BQ    16            // batch
#define ENC   512
#define ATT   512
#define DEC   512
#define NMEL  80

// ---------------- device math helpers ----------------
__device__ __forceinline__ float sigf(float x) { return 1.0f / (1.0f + __expf(-x)); }
__device__ __forceinline__ float tanh_(float x) {
  float c = fminf(fmaxf(x, -15.f), 15.f);
  float e = __expf(2.f * c);
  return (e - 1.f) / (e + 1.f);
}
__device__ __forceinline__ float softplus_(float x) {
  return x > 15.f ? x : __logf(1.f + __expf(x));
}
__device__ __forceinline__ unsigned wanghash(unsigned s) {
  s = (s ^ 61u) ^ (s >> 16); s *= 9u; s ^= s >> 4; s *= 0x27d4eb2du; s ^= s >> 15;
  return s;
}

// ---------------- WMMA fragment loaders ----------------
// A tile: 16x32 f16, row-major with row stride `ld` halfs.
// ISA 16-bit A layout: lanes 0-15 hold K=(0..7,16..23), lanes 16-31 hold
// K=(8..15,24..31) for row M=lane&15.
__device__ __forceinline__ v16h load_a(const _Float16* base, int ld, int kt) {
  int lane = threadIdx.x & 31;
  const _Float16* p = base + (size_t)(lane & 15) * ld + (kt << 5) + ((lane >> 4) << 3);
  h8 lo = *(const h8*)(p);
  h8 hi = *(const h8*)(p + 16);
  return __builtin_shufflevector(lo, hi, 0,1,2,3,4,5,6,7,8,9,10,11,12,13,14,15);
}

// B tile: 32x16 f16 pre-swizzled: tile = 512 contiguous halfs,
// lane L owns halfs [L*16, L*16+16) = B[k = (L>>4)*8 + (i&7) + (i>>3)*16][n = L&15].
__device__ __forceinline__ v16h load_b(const _Float16* tiles, int tileIdx) {
  int lane = threadIdx.x & 31;
  const _Float16* p = tiles + ((size_t)tileIdx << 9) + (lane << 4);
  h8 lo = *(const h8*)(p);
  h8 hi = *(const h8*)(p + 8);
  return __builtin_shufflevector(lo, hi, 0,1,2,3,4,5,6,7,8,9,10,11,12,13,14,15);
}

__device__ __forceinline__ v8f wmma_acc(v16h a, v16h b, v8f c) {
  return __builtin_amdgcn_wmma_f32_16x16x32_f16(false, a, false, b, (short)0, c, false, false);
}

// D (16x16 f32): VGPR r -> M = r + 8*(lane>>4), N = lane&15.
__device__ __forceinline__ void store_d(float* G, int ldn, int nbase, v8f d) {
  int lane = threadIdx.x & 31;
  int n = nbase + (lane & 15);
  int m0 = (lane >> 4) * 8;
#pragma unroll
  for (int r = 0; r < 8; ++r) G[(size_t)(m0 + r) * ldn + n] = d[r];
}

// Dual independent accumulators: even K-tiles -> a0, odd -> a1. Breaks the
// WMMA D->C serial chain so consecutive v_wmma ops co-execute (no v_nop pad).
struct AccPair { v8f a0, a1; };
__device__ __forceinline__ AccPair acc_zero() {
  AccPair ac;
  ac.a0 = (v8f){0.f,0.f,0.f,0.f,0.f,0.f,0.f,0.f};
  ac.a1 = (v8f){0.f,0.f,0.f,0.f,0.f,0.f,0.f,0.f};
  return ac;
}

// Branch-free K-region sub-loop: nkt (even) tiles of A from one buffer.
__device__ __forceinline__ void gemm_region(AccPair& ac, const _Float16* __restrict__ abase,
                                            int ld, const _Float16* __restrict__ btiles,
                                            int nkt) {
  for (int kt = 0; kt < nkt; kt += 2) {
    __builtin_prefetch(btiles + (((size_t)(kt + 2)) << 9), 0, 3);
    ac.a0 = wmma_acc(load_a(abase, ld, kt),     load_b(btiles, kt),     ac.a0);
    ac.a1 = wmma_acc(load_a(abase, ld, kt + 1), load_b(btiles, kt + 1), ac.a1);
  }
}

// ---------------- grid barrier (monotonic counter, reset by init kernel) ----
__device__ __forceinline__ void gbar(unsigned* ctr, int& epoch) {
  __syncthreads();
  if (threadIdx.x == 0) {
    __threadfence();
    atomicAdd(ctr, 1u);
    unsigned tgt = (unsigned)(++epoch) * (unsigned)NWG;
    while (__hip_atomic_load(ctr, __ATOMIC_ACQUIRE, __HIP_MEMORY_SCOPE_AGENT) < tgt) {
      __builtin_amdgcn_s_sleep(2);
    }
  }
  __syncthreads();
}

// ---------------- prep kernels ----------------
__global__ void zero_ws_kernel(unsigned* p, unsigned nwords) {
  for (unsigned i = blockIdx.x * blockDim.x + threadIdx.x; i < nwords;
       i += gridDim.x * blockDim.x)
    p[i] = 0u;
}

// Convert torch-style W [N][Ksrc] (f32, row-major) into f16 B-fragment tiles.
__global__ void convert_b_kernel(const float* __restrict__ W, _Float16* __restrict__ dst,
                                 int N, int Ksrc, int NT, int dstKT, int kTileOff) {
  int total = (Ksrc >> 5) * NT * 512;
  int idx = blockIdx.x * blockDim.x + threadIdx.x;
  if (idx >= total) return;
  int e = idx & 511;
  int tile = idx >> 9;
  int KTs = Ksrc >> 5;
  int ktl = tile % KTs;
  int nt  = tile / KTs;
  int laneL = e >> 4;
  int i = e & 15;
  int k = ktl * 32 + ((laneL >> 4) << 3) + (i & 7) + ((i >> 3) << 4);
  int n = nt * 16 + (laneL & 15);
  float v = (n < N) ? W[(size_t)n * Ksrc + k] : 0.0f;
  dst[(((size_t)(nt * dstKT + kTileOff + ktl)) << 9) + e] = (_Float16)v;
}

// Prenet for all TDEC frames (dropout via counter hash; structure/cost match).
__global__ void prenet_kernel(const float* __restrict__ mel,
                              const float* __restrict__ w1,
                              const float* __restrict__ w2,
                              _Float16* __restrict__ out) {
  __shared__ float din[BQ * NMEL];
  __shared__ float x1[BQ * 256];
  int s = blockIdx.x;
  int tid = threadIdx.x;
  for (int idx = tid; idx < BQ * NMEL; idx += NTHR) {
    int b = idx / NMEL, c = idx - b * NMEL;
    float v = 0.f;
    if (s > 0) v = mel[(size_t)b * 1200 * NMEL + (size_t)(2 * (s - 1) + 1) * NMEL + c];
    din[idx] = v;
  }
  __syncthreads();
  for (int idx = tid; idx < BQ * 256; idx += NTHR) {
    int b = idx >> 8, n = idx & 255;
    float acc = 0.f;
#pragma unroll 4
    for (int k = 0; k < NMEL; ++k) acc += din[b * NMEL + k] * w1[(size_t)n * NMEL + k];
    acc = fmaxf(acc, 0.f);
    acc = (wanghash(0x1234567u ^ (s * 4096 + idx)) & 1u) ? acc * 2.f : 0.f;
    x1[idx] = acc;
  }
  __syncthreads();
  for (int idx = tid; idx < BQ * 256; idx += NTHR) {
    int b = idx >> 8, n = idx & 255;
    float acc = 0.f;
#pragma unroll 4
    for (int k = 0; k < 256; ++k) acc += x1[b * 256 + k] * w2[(size_t)n * 256 + k];
    acc = fmaxf(acc, 0.f);
    acc = (wanghash(0x9E3779B9u ^ (s * 4096 + idx)) & 1u) ? acc * 2.f : 0.f;
    out[(size_t)s * 4096 + idx] = (_Float16)acc;
  }
}

// ---------------- persistent decoder ----------------
struct DecCtx {
  const float* __restrict__ memory;       // [16][600][512]
  const float* __restrict__ mpitch;       // [16][600][512]
  const int*   __restrict__ memlen;       // [16]
  const float* __restrict__ att_bih; const float* __restrict__ att_bhh;
  const float* __restrict__ q_b1;    const float* __restrict__ q_b2;
  const float* __restrict__ dec_bih; const float* __restrict__ dec_bhh;
  const float* __restrict__ proj_b;
  const _Float16* __restrict__ WtAtt; const _Float16* __restrict__ WtDec;
  const _Float16* __restrict__ WtQ1;  const _Float16* __restrict__ WtQ2;
  const _Float16* __restrict__ WtProj;
  const _Float16* __restrict__ prenet;    // [600][16][256] f16
  _Float16* __restrict__ hA0; _Float16* __restrict__ hA1;
  _Float16* __restrict__ hD0; _Float16* __restrict__ hD1;
  _Float16* __restrict__ ctx0; _Float16* __restrict__ ctx1;
  _Float16* __restrict__ ctxp0; _Float16* __restrict__ ctxp1;
  _Float16* __restrict__ q1h;
  float* __restrict__ cA; float* __restrict__ cD;
  float* __restrict__ G;  float* __restrict__ alpha;
  float* __restrict__ qbuf; float* __restrict__ wS;
  float* __restrict__ sS;   float* __restrict__ muS;
  unsigned* __restrict__ bar;
  float* __restrict__ out_mel;            // [16][1200][80]
  float* __restrict__ out_align;          // [16][600][600]
};

__global__ __launch_bounds__(NTHR, 1) void decoder_kernel(DecCtx cx) {
  const int tid  = threadIdx.x;
  const int lane = tid & 31;
  const int wid  = tid >> 5;
  const int gw   = blockIdx.x * (NTHR / 32) + wid;   // 0..127 global wave
  const int gtid = blockIdx.x * NTHR + tid;
  int epoch = 0;

  for (int t = 0; t < TDEC; ++t) {
    const int p = t & 1;
    _Float16* hAr  = p ? cx.hA1  : cx.hA0;   // read (h_att  t-1)
    _Float16* hAw  = p ? cx.hA0  : cx.hA1;   // write (h_att t)
    _Float16* hDr  = p ? cx.hD1  : cx.hD0;
    _Float16* hDw  = p ? cx.hD0  : cx.hD1;
    _Float16* ctxR = p ? cx.ctx1 : cx.ctx0;  // ctx from step t-1
    _Float16* ctxW = p ? cx.ctx0 : cx.ctx1;  // ctx computed at step t
    _Float16* pctW = p ? cx.ctxp0 : cx.ctxp1;
    const _Float16* pre_t = cx.prenet + (size_t)t * 4096;

    // ---- P1: attention-LSTM gates [16x1280]x[1280x2048]; branch-free
    //      K-region sub-loops: prenet(8) | ctx(16) | hA(16) ----------------
    {
      const _Float16* bt = cx.WtAtt + ((size_t)(gw * 40) << 9);
      AccPair ac = acc_zero();
      gemm_region(ac, pre_t, 256, bt,                     8);
      gemm_region(ac, ctxR,  512, bt + ((size_t)8  << 9), 16);
      gemm_region(ac, hAr,   512, bt + ((size_t)24 << 9), 16);
      store_d(cx.G, 2048, gw * 16, ac.a0 + ac.a1);
    }
    gbar(cx.bar, epoch);

    // ---- P2: att-LSTM pointwise (i,f,g,o -> h,c) ---------------------------
    for (int idx = gtid; idx < BQ * ATT; idx += NTOT) {
      int b = idx >> 9, j = idx & 511;
      const float* Gb = cx.G + (size_t)b * 2048;
      float gi = Gb[j]        + cx.att_bih[j]        + cx.att_bhh[j];
      float gf = Gb[512 + j]  + cx.att_bih[512 + j]  + cx.att_bhh[512 + j];
      float gg = Gb[1024 + j] + cx.att_bih[1024 + j] + cx.att_bhh[1024 + j];
      float go = Gb[1536 + j] + cx.att_bih[1536 + j] + cx.att_bhh[1536 + j];
      float c2 = sigf(gf) * cx.cA[idx] + sigf(gi) * tanh_(gg);
      cx.cA[idx] = c2;
      hAw[idx] = (_Float16)(sigf(go) * tanh_(c2));
    }
    gbar(cx.bar, epoch);

    // ---- P3: q1 = tanh(hA @ q_w1^T + b1): waves 0..15 ---------------------
    if (gw < 16) {
      AccPair ac = acc_zero();
      gemm_region(ac, hAw, 512, cx.WtQ1 + ((size_t)(gw * 16) << 9), 16);
      v8f acc = ac.a0 + ac.a1;
      int n = gw * 16 + (lane & 15);
      int m0 = (lane >> 4) * 8;
      float bq = cx.q_b1[n];
#pragma unroll
      for (int r = 0; r < 8; ++r)
        cx.q1h[(size_t)(m0 + r) * 256 + n] = (_Float16)tanh_(acc[r] + bq);
    }
    gbar(cx.bar, epoch);

    // ---- P4: q2 = q1 @ q_w2^T + b2 (15 cols, padded tile): wave 0 ---------
    if (gw == 0) {
      AccPair ac = acc_zero();
      gemm_region(ac, cx.q1h, 256, cx.WtQ2, 8);
      v8f acc = ac.a0 + ac.a1;
      int n = lane & 15;
      int m0 = (lane >> 4) * 8;
      float bq = (n < 15) ? cx.q_b2[n] : 0.f;
#pragma unroll
      for (int r = 0; r < 8; ++r) cx.qbuf[(m0 + r) * 16 + n] = acc[r] + bq;
    }
    gbar(cx.bar, epoch);

    // ---- P5: mixture params: w=softmax+1e-6, sigma, mu += softplus(delta) --
    if (blockIdx.x == 0 && tid < BQ) {
      const float* q = cx.qbuf + tid * 16;
      float mx = q[0];
#pragma unroll
      for (int m = 1; m < 5; ++m) mx = fmaxf(mx, q[m]);
      float e[5], s = 0.f;
#pragma unroll
      for (int m = 0; m < 5; ++m) { e[m] = __expf(q[m] - mx); s += e[m]; }
#pragma unroll
      for (int m = 0; m < 5; ++m) {
        cx.wS[tid * 5 + m] = e[m] / s + 1e-6f;
        cx.sS[tid * 5 + m] = softplus_(q[5 + m]) + 0.5f;
        float mu = cx.muS[tid * 5 + m] + softplus_(q[10 + m]);
        cx.muS[tid * 5 + m] = mu;
      }
    }
    gbar(cx.bar, epoch);

    // ---- P6: alpha[b][te] over 16x600, masked; also -> alignments output ---
    for (int idx = gtid; idx < BQ * TENC; idx += NTOT) {
      int b = idx / TENC, te = idx - b * TENC;
      float a = 0.f;
      if (te < cx.memlen[b]) {
        float jj = (float)te + 0.5f;
#pragma unroll
        for (int m = 0; m < 5; ++m) {
          float sg = cx.sS[b * 5 + m];
          float zc = (jj - cx.muS[b * 5 + m]) / sg;
          float hh = 0.5f / sg;
          a += cx.wS[b * 5 + m] * (sigf(zc + hh) - sigf(zc - hh));
        }
      }
      cx.alpha[idx] = a;
      cx.out_align[(size_t)b * (TDEC * TENC) + (size_t)t * TENC + te] = a;
    }
    gbar(cx.bar, epoch);

    // ---- P7: ctx/ctx_p = einsum('bt,btd->bd'): float4 streaming from L2 ----
    // 128 tasks: tensor(2) x col32-tile(16) x batch-quad(4).
    {
      const float* src = (gw & 1) ? cx.mpitch : cx.memory;
      _Float16* dst = (gw & 1) ? pctW : ctxW;
      int ct = (gw >> 1) & 15;
      int bqd = gw >> 5;
      int b = bqd * 4 + (lane >> 3);
      int col4 = ct * 32 + (lane & 7) * 4;
      const float4* mb = (const float4*)(src + (size_t)b * TENC * ENC + col4);
      const float* ab = cx.alpha + b * TENC;
      float ax = 0.f, ay = 0.f, az = 0.f, aw = 0.f;
#pragma unroll 4
      for (int te = 0; te < TENC; ++te) {
        float al = ab[te];
        float4 v = mb[(size_t)te * (ENC / 4)];
        ax += al * v.x; ay += al * v.y; az += al * v.z; aw += al * v.w;
      }
      h4 hv = { (_Float16)ax, (_Float16)ay, (_Float16)az, (_Float16)aw };
      *(h4*)(dst + (size_t)b * ENC + col4) = hv;
    }
    gbar(cx.bar, epoch);

    // ---- P8: dec-LSTM gates [16x2048]x[2048x2048]; regions hA|ctx|ctxp|hD --
    {
      const _Float16* bt = cx.WtDec + ((size_t)(gw * 64) << 9);
      AccPair ac = acc_zero();
      gemm_region(ac, hAw,  512, bt,                     16);
      gemm_region(ac, ctxW, 512, bt + ((size_t)16 << 9), 16);
      gemm_region(ac, pctW, 512, bt + ((size_t)32 << 9), 16);
      gemm_region(ac, hDr,  512, bt + ((size_t)48 << 9), 16);
      store_d(cx.G, 2048, gw * 16, ac.a0 + ac.a1);
    }
    gbar(cx.bar, epoch);

    // ---- P9: dec-LSTM pointwise -------------------------------------------
    for (int idx = gtid; idx < BQ * DEC; idx += NTOT) {
      int b = idx >> 9, j = idx & 511;
      const float* Gb = cx.G + (size_t)b * 2048;
      float gi = Gb[j]        + cx.dec_bih[j]        + cx.dec_bhh[j];
      float gf = Gb[512 + j]  + cx.dec_bih[512 + j]  + cx.dec_bhh[512 + j];
      float gg = Gb[1024 + j] + cx.dec_bih[1024 + j] + cx.dec_bhh[1024 + j];
      float go = Gb[1536 + j] + cx.dec_bih[1536 + j] + cx.dec_bhh[1536 + j];
      float c2 = sigf(gf) * cx.cD[idx] + sigf(gi) * tanh_(gg);
      cx.cD[idx] = c2;
      hDw[idx] = (_Float16)(sigf(go) * tanh_(c2));
    }
    gbar(cx.bar, epoch);

    // ---- P10: projection [16x1536]x[1536x160] -> mel output ---------------
    if (gw < 10) {
      const _Float16* bt = cx.WtProj + ((size_t)(gw * 48) << 9);
      AccPair ac = acc_zero();
      gemm_region(ac, hDw,  512, bt,                     16);
      gemm_region(ac, ctxW, 512, bt + ((size_t)16 << 9), 16);
      gemm_region(ac, pctW, 512, bt + ((size_t)32 << 9), 16);
      v8f acc = ac.a0 + ac.a1;
      int n = gw * 16 + (lane & 15);
      int m0 = (lane >> 4) * 8;
      float pb = cx.proj_b[n];
#pragma unroll
      for (int r = 0; r < 8; ++r) {
        int b = m0 + r;
        cx.out_mel[(size_t)b * (TDEC * NMEL * 2) + (size_t)t * 160 + n] = acc[r] + pb;
      }
    }
    gbar(cx.bar, epoch);
  }
}

// ---------------------------------------------------------------------------
extern "C" void kernel_launch(void* const* d_in, const int* in_sizes, int n_in,
                              void* d_out, int out_size, void* d_ws, size_t ws_size,
                              hipStream_t stream) {
  (void)in_sizes; (void)n_in; (void)out_size; (void)ws_size;
  const float* memory  = (const float*)d_in[0];
  const float* mel     = (const float*)d_in[1];
  const int*   memlen  = (const int*)d_in[2];
  const float* mpitch  = (const float*)d_in[3];
  const float* pw1     = (const float*)d_in[4];
  const float* pw2     = (const float*)d_in[5];
  const float* att_wih = (const float*)d_in[6];
  const float* att_whh = (const float*)d_in[7];
  const float* att_bih = (const float*)d_in[8];
  const float* att_bhh = (const float*)d_in[9];
  const float* q_w1    = (const float*)d_in[10];
  const float* q_b1    = (const float*)d_in[11];
  const float* q_w2    = (const float*)d_in[12];
  const float* q_b2    = (const float*)d_in[13];
  const float* dec_wih = (const float*)d_in[14];
  const float* dec_whh = (const float*)d_in[15];
  const float* dec_bih = (const float*)d_in[16];
  const float* dec_bhh = (const float*)d_in[17];
  const float* proj_w  = (const float*)d_in[18];
  const float* proj_b  = (const float*)d_in[19];

  char* ws = (char*)d_ws;
  size_t o = 0;
  auto alloc = [&](size_t bytes) { size_t r = o; o = (o + bytes + 255) & ~(size_t)255; return r; };

  // ---- zero-initialized state block (contiguous, reset every launch) ----
  size_t oBar   = alloc(256);
  size_t oCA    = alloc(8192 * 4);
  size_t oCD    = alloc(8192 * 4);
  size_t oHA0   = alloc(8192 * 2);
  size_t oHD0   = alloc(8192 * 2);
  size_t oCtx0  = alloc(8192 * 2);
  size_t oCtxp0 = alloc(8192 * 2);
  size_t oMu    = alloc(80 * 4);
  size_t zeroEnd = o;
  // ---- non-zeroed scratch ----
  size_t oHA1   = alloc(8192 * 2);
  size_t oHD1   = alloc(8192 * 2);
  size_t oCtx1  = alloc(8192 * 2);
  size_t oCtxp1 = alloc(8192 * 2);
  size_t oG     = alloc(16 * 2048 * 4);
  size_t oAlpha = alloc(16 * 600 * 4);
  size_t oQbuf  = alloc(16 * 16 * 4);
  size_t oQ1h   = alloc(16 * 256 * 2);
  size_t oWw    = alloc(80 * 4);
  size_t oSw    = alloc(80 * 4);
  size_t oPre   = alloc((size_t)600 * 16 * 256 * 2);       // 4.9 MB
  size_t oWtAtt = alloc((size_t)128 * 40 * 512 * 2);       // 5.24 MB
  size_t oWtDec = alloc((size_t)128 * 64 * 512 * 2);       // 8.39 MB
  size_t oWtQ1  = alloc((size_t)16 * 16 * 512 * 2);
  size_t oWtQ2  = alloc((size_t)1 * 8 * 512 * 2);
  size_t oWtPj  = alloc((size_t)10 * 48 * 512 * 2);

  // 1) zero state
  unsigned zwords = (unsigned)(zeroEnd / 4);
  zero_ws_kernel<<<(zwords + 255) / 256, 256, 0, stream>>>((unsigned*)ws, zwords);

  // 2) weight conversion to f16 B-fragment tiles (combined [wih|whh] along K)
  auto conv = [&](const float* W, size_t dstOff, int N, int K, int NT, int dstKT, int koff) {
    int total = (K >> 5) * NT * 512;
    convert_b_kernel<<<(total + 255) / 256, 256, 0, stream>>>(
        W, (_Float16*)(ws + dstOff), N, K, NT, dstKT, koff);
  };
  conv(att_wih, oWtAtt, 2048, 768, 128, 40, 0);
  conv(att_whh, oWtAtt, 2048, 512, 128, 40, 24);
  conv(dec_wih, oWtDec, 2048, 1536, 128, 64, 0);
  conv(dec_whh, oWtDec, 2048, 512, 128, 64, 48);
  conv(q_w1,    oWtQ1,  256,  512,  16,  16, 0);
  conv(q_w2,    oWtQ2,  15,   256,  1,   8,  0);
  conv(proj_w,  oWtPj,  160,  1536, 10,  48, 0);

  // 3) prenet over all 600 decoder frames
  prenet_kernel<<<600, NTHR, 0, stream>>>(mel, pw1, pw2, (_Float16*)(ws + oPre));

  // 4) persistent decoder scan
  DecCtx cx;
  cx.memory = memory; cx.mpitch = mpitch; cx.memlen = memlen;
  cx.att_bih = att_bih; cx.att_bhh = att_bhh;
  cx.q_b1 = q_b1; cx.q_b2 = q_b2;
  cx.dec_bih = dec_bih; cx.dec_bhh = dec_bhh; cx.proj_b = proj_b;
  cx.WtAtt = (const _Float16*)(ws + oWtAtt);
  cx.WtDec = (const _Float16*)(ws + oWtDec);
  cx.WtQ1  = (const _Float16*)(ws + oWtQ1);
  cx.WtQ2  = (const _Float16*)(ws + oWtQ2);
  cx.WtProj= (const _Float16*)(ws + oWtPj);
  cx.prenet= (const _Float16*)(ws + oPre);
  cx.hA0 = (_Float16*)(ws + oHA0);   cx.hA1 = (_Float16*)(ws + oHA1);
  cx.hD0 = (_Float16*)(ws + oHD0);   cx.hD1 = (_Float16*)(ws + oHD1);
  cx.ctx0 = (_Float16*)(ws + oCtx0); cx.ctx1 = (_Float16*)(ws + oCtx1);
  cx.ctxp0 = (_Float16*)(ws + oCtxp0); cx.ctxp1 = (_Float16*)(ws + oCtxp1);
  cx.q1h = (_Float16*)(ws + oQ1h);
  cx.cA = (float*)(ws + oCA); cx.cD = (float*)(ws + oCD);
  cx.G = (float*)(ws + oG); cx.alpha = (float*)(ws + oAlpha);
  cx.qbuf = (float*)(ws + oQbuf);
  cx.wS = (float*)(ws + oWw); cx.sS = (float*)(ws + oSw); cx.muS = (float*)(ws + oMu);
  cx.bar = (unsigned*)(ws + oBar);
  cx.out_mel = (float*)d_out;
  cx.out_align = (float*)d_out + (size_t)16 * 1200 * 80;

  decoder_kernel<<<NWG, NTHR, 0, stream>>>(cx);
}
